// MultiHeadSelfAttention_83107617178178
// MI455X (gfx1250) — matrix operand order repaired and checked by reading, not compiled, not run on hardware
//
#include <hip/hip_runtime.h>

typedef __attribute__((ext_vector_type(16))) __bf16 v16bf;
typedef __attribute__((ext_vector_type(8)))  float  v8f;
typedef __attribute__((ext_vector_type(4)))  int    v4i;

#define DM    1024
#define SEQ   2048
#define BATCH 2
#define NH    16
#define HD    64
#define ROWS  (BATCH * SEQ)   // 4096

#if defined(__AMDGCN__) && __has_builtin(__builtin_amdgcn_global_load_async_to_lds_b128)
#define HAS_ASYNC_LDS 1
#else
#define HAS_ASYNC_LDS 0
#endif

static __device__ __forceinline__ unsigned short f2bf(float f) {
  unsigned int u = __float_as_uint(f);
  u += 0x7fffu + ((u >> 16) & 1u);   // round-to-nearest-even
  return (unsigned short)(u >> 16);
}

union AFrag { v16bf v; uint4 q[2]; };

static __device__ __forceinline__ v8f wmma_bf16(v16bf a, v16bf b, v8f c) {
  return __builtin_amdgcn_wmma_f32_16x16x32_bf16(false, a, false, b, (short)0, c,
                                                 false, false);
}

// 16-byte global -> LDS copy: async path on CDNA5, sync fallback otherwise.
static __device__ __forceinline__ void copy16_to_lds(const unsigned short* g,
                                                     unsigned short* l) {
#if HAS_ASYNC_LDS
  __builtin_amdgcn_global_load_async_to_lds_b128(
      (__attribute__((address_space(1))) v4i*)g,
      (__attribute__((address_space(3))) v4i*)l, 0, 0);
#else
  *(uint4*)l = *(const uint4*)g;
#endif
}

static __device__ __forceinline__ void wait_async_copies() {
#if HAS_ASYNC_LDS
  asm volatile("s_wait_asynccnt 0" ::: "memory");
#endif
}

// -------------------- LayerNorm (f32 -> bf16) --------------------
__global__ __launch_bounds__(256)
void ln_kernel(const float* __restrict__ x, const float* __restrict__ gamma,
               const float* __restrict__ beta, unsigned short* __restrict__ xn) {
  const int row = blockIdx.x;
  const float* xr = x + (size_t)row * DM;
  __shared__ float red[256];
  const int tid = threadIdx.x;

  float s = 0.f;
  for (int i = tid; i < DM; i += 256) s += xr[i];
  red[tid] = s; __syncthreads();
  for (int o = 128; o > 0; o >>= 1) { if (tid < o) red[tid] += red[tid + o]; __syncthreads(); }
  const float mu = red[0] * (1.0f / DM);
  __syncthreads();

  float v = 0.f;
  for (int i = tid; i < DM; i += 256) { float d = xr[i] - mu; v += d * d; }
  red[tid] = v; __syncthreads();
  for (int o = 128; o > 0; o >>= 1) { if (tid < o) red[tid] += red[tid + o]; __syncthreads(); }
  const float rstd = rsqrtf(red[0] * (1.0f / DM) + 1e-5f);

  for (int i = tid; i < DM; i += 256)
    xn[(size_t)row * DM + i] = f2bf((xr[i] - mu) * rstd * gamma[i] + beta[i]);
}

// ---------- Weight transpose: W[k][n] f32 -> Wt[n][k] bf16 ----------
__global__ __launch_bounds__(256)
void wtrans_kernel(const float* __restrict__ W, unsigned short* __restrict__ Wt) {
  const int idx = blockIdx.x * 256 + threadIdx.x;   // over DM*DM
  const int n = idx >> 10;
  const int k = idx & (DM - 1);
  Wt[idx] = f2bf(W[(size_t)k * DM + n]);
}

// ---------- V transpose: V[(b*S+s)][d] bf16 -> Vt[b][d][s] bf16 ----------
__global__ __launch_bounds__(256)
void vtrans_kernel(const unsigned short* __restrict__ V, unsigned short* __restrict__ Vt) {
  const int idx = blockIdx.x * 256 + threadIdx.x;   // over BATCH*DM*SEQ = 2^22
  const int s = idx & (SEQ - 1);
  const int d = (idx >> 11) & (DM - 1);
  const int b = idx >> 21;
  Vt[idx] = V[((size_t)(b * SEQ + s)) * DM + d];
}

// -------------------- WMMA GEMM with async-LDS staging --------------------
// C = alpha*(A @ Bt^T + bias).  A: [M,1024] bf16 row-major.  Bt: [1024,1024] bf16,
// Bt[n][k].  Block = 256 threads (8 waves) computes a 128x64 tile; per K-step of 32
// the block stages A(128x32) and B(64x32) in LDS via async global->LDS copies.
template<bool F32OUT>
__global__ __launch_bounds__(256)
void gemm_wmma(const unsigned short* __restrict__ A,
               const unsigned short* __restrict__ Bt,
               const float* __restrict__ bias,
               void* __restrict__ Cout, float alpha) {
  __shared__ unsigned short lA[128 * 32];   // 8 KiB
  __shared__ unsigned short lB[64 * 32];    // 4 KiB

  const int tid  = threadIdx.x;
  const int lane = tid & 31;
  const int wave = tid >> 5;                 // 0..7 -> M sub-tile
  const int m0   = blockIdx.y * 128;
  const int n0   = blockIdx.x * 64;

  const int am  = lane & 15;                 // A frag: M index
  const int akb = (lane < 16) ? 0 : 16;      // A frag: K base
  const int bn  = lane & 15;                 // B frag: N index
  const int bkb = (lane < 16) ? 0 : 16;      // B frag: K base

  v8f acc[4] = {};

  for (int k0 = 0; k0 < DM; k0 += 32) {
    // ---- stage tiles: A = 512 chunks of 16B (2/thread), B = 256 chunks (1/thread)
    {
      int r = tid >> 2, c = tid & 3;
      copy16_to_lds(A + (size_t)(m0 + r) * DM + k0 + c * 8, &lA[r * 32 + c * 8]);
      int t2 = tid + 256;
      r = t2 >> 2; c = t2 & 3;
      copy16_to_lds(A + (size_t)(m0 + r) * DM + k0 + c * 8, &lA[r * 32 + c * 8]);
      r = tid >> 2; c = tid & 3;
      copy16_to_lds(Bt + (size_t)(n0 + r) * DM + k0 + c * 8, &lB[r * 32 + c * 8]);
    }
    wait_async_copies();
    __syncthreads();

    AFrag a;
    const unsigned short* ar = &lA[(wave * 16 + am) * 32];
    a.q[0] = *(const uint4*)(ar + akb);
    a.q[1] = *(const uint4*)(ar + 8 + akb);
#pragma unroll
    for (int t = 0; t < 4; ++t) {
      const unsigned short* br = &lB[(t * 16 + bn) * 32 + bkb];
      AFrag b;
      b.q[0] = *(const uint4*)(br);
      b.q[1] = *(const uint4*)(br + 8);
      acc[t] = wmma_bf16(a.v, b.v, acc[t]);
    }
    __syncthreads();
  }

  const int m0w = m0 + wave * 16;
  const int cn  = lane & 15;
  const int chi = (lane >> 4) << 3;
#pragma unroll
  for (int t = 0; t < 4; ++t) {
    const int gn = n0 + t * 16 + cn;
    const float bv = bias ? bias[gn] : 0.f;
#pragma unroll
    for (int r = 0; r < 8; ++r) {
      const int gm = m0w + chi + r;
      const float val = alpha * (acc[t][r] + bv);
      if (F32OUT) ((float*)Cout)[(size_t)gm * DM + gn] = val;
      else        ((unsigned short*)Cout)[(size_t)gm * DM + gn] = f2bf(val);
    }
  }
}

// -------------------- Flash attention (one wave per 32-query tile) --------------------
// Q,K: bf16 [B*S, D] (Q pre-scaled by 1/8).  Vt: bf16 [B, D, S].  O: bf16 [B*S, D].
// Two 16-row query tiles per wave share all K/V fragments (halves L2 traffic).
// Row-sums via P @ ones WMMA; cross-lane max uses a batched butterfly (4 waits).
__global__ __launch_bounds__(32)
void attn_kernel(const unsigned short* __restrict__ Q,
                 const unsigned short* __restrict__ Km,
                 const unsigned short* __restrict__ Vt,
                 unsigned short* __restrict__ O) {
  __shared__ unsigned short pbuf[2 * 16 * 32];   // P staging, one 16x32 tile per q-tile
  const int lane = threadIdx.x;
  const int qt = blockIdx.x, h = blockIdx.y, b = blockIdx.z;

  const int am  = lane & 15;
  const int akb = (lane < 16) ? 0 : 16;
  const int bn  = lane & 15;
  const int bkb = (lane < 16) ? 0 : 16;
  const int cn  = lane & 15;
  const int chi = (lane >> 4) << 3;

  // Q A-fragments for two 16-row tiles, each 16x64 as two 16x32 chunks
  AFrag aq[2][2];
#pragma unroll
  for (int u = 0; u < 2; ++u) {
    const unsigned short* qrow =
        Q + (size_t)(b * SEQ + qt * 32 + u * 16 + am) * DM + h * HD;
#pragma unroll
    for (int c = 0; c < 2; ++c) {
      aq[u][c].q[0] = *(const uint4*)(qrow + c * 32 + akb);
      aq[u][c].q[1] = *(const uint4*)(qrow + c * 32 + 8 + akb);
    }
  }

  // all-ones bf16 B fragment (every column of P@ones = row-sum of P)
  AFrag ones;
#pragma unroll
  for (int i = 0; i < 2; ++i) {
    ones.q[i].x = 0x3F803F80u; ones.q[i].y = 0x3F803F80u;
    ones.q[i].z = 0x3F803F80u; ones.q[i].w = 0x3F803F80u;
  }

  float rm[2][8];
#pragma unroll
  for (int u = 0; u < 2; ++u)
#pragma unroll
    for (int r = 0; r < 8; ++r) rm[u][r] = -3.0e38f;
  v8f acc[2][4] = {};
  v8f accs[2] = {};

  for (int j = 0; j < SEQ; j += 32) {
    __builtin_prefetch((const void*)(Km + (size_t)(b * SEQ + ((j + 32) & (SEQ - 1)) + bn) * DM + h * HD), 0, 2);

    // ---- load all K fragments (2 key tiles x 2 d-chunks) and V fragments ----
    AFrag bk[2][2];
#pragma unroll
    for (int t = 0; t < 2; ++t) {
      const unsigned short* krow =
          Km + (size_t)(b * SEQ + j + t * 16 + bn) * DM + h * HD;
      bk[t][0].q[0] = *(const uint4*)(krow + bkb);
      bk[t][0].q[1] = *(const uint4*)(krow + bkb + 8);
      bk[t][1].q[0] = *(const uint4*)(krow + 32 + bkb);
      bk[t][1].q[1] = *(const uint4*)(krow + 32 + bkb + 8);
    }
    AFrag bv[4];
#pragma unroll
    for (int t = 0; t < 4; ++t) {
      const unsigned short* vrow =
          Vt + ((size_t)b * DM + h * HD + t * 16 + bn) * SEQ + j + bkb;
      bv[t].q[0] = *(const uint4*)(vrow);
      bv[t].q[1] = *(const uint4*)(vrow + 8);
    }

    // ---- scores for both q-tiles ----
    v8f sc[2][2];
#pragma unroll
    for (int u = 0; u < 2; ++u)
#pragma unroll
      for (int t = 0; t < 2; ++t) {
        v8f z = {};
        z = wmma_bf16(aq[u][0].v, bk[t][0].v, z);
        z = wmma_bf16(aq[u][1].v, bk[t][1].v, z);
        sc[u][t] = z;
      }

    // ---- online softmax per q-tile ----
#pragma unroll
    for (int u = 0; u < 2; ++u) {
      float mx[8];
#pragma unroll
      for (int r = 0; r < 8; ++r) mx[r] = fmaxf(sc[u][0][r], sc[u][1][r]);
      // batched butterfly: 8 independent shuffles per step -> one wait per step
#pragma unroll
      for (int o = 1; o < 16; o <<= 1) {
        float tv[8];
#pragma unroll
        for (int r = 0; r < 8; ++r) tv[r] = __shfl_xor(mx[r], o);
#pragma unroll
        for (int r = 0; r < 8; ++r) mx[r] = fmaxf(mx[r], tv[r]);
      }
      float f[8];
#pragma unroll
      for (int r = 0; r < 8; ++r) {
        const float mnew = fmaxf(rm[u][r], mx[r]);
        sc[u][0][r] = __expf(sc[u][0][r] - mnew);
        sc[u][1][r] = __expf(sc[u][1][r] - mnew);
        f[r] = __expf(rm[u][r] - mnew);
        rm[u][r] = mnew;
      }
#pragma unroll
      for (int t = 0; t < 4; ++t)
#pragma unroll
        for (int r = 0; r < 8; ++r) acc[u][t][r] = acc[u][t][r] * f[r];
#pragma unroll
      for (int r = 0; r < 8; ++r) accs[u][r] = accs[u][r] * f[r];

      // P tile u: C-layout -> LDS (bf16)
#pragma unroll
      for (int t = 0; t < 2; ++t)
#pragma unroll
        for (int r = 0; r < 8; ++r)
          pbuf[u * 512 + (chi + r) * 32 + t * 16 + cn] = f2bf(sc[u][t][r]);
    }
    asm volatile("s_wait_dscnt 0" ::: "memory");

    // ---- read P in A-layout, accumulate row-sums and P @ V ----
#pragma unroll
    for (int u = 0; u < 2; ++u) {
      AFrag pa;
      pa.q[0] = *(const uint4*)(&pbuf[u * 512 + am * 32 + akb]);
      pa.q[1] = *(const uint4*)(&pbuf[u * 512 + am * 32 + 8 + akb]);
      accs[u] = wmma_bf16(pa.v, ones.v, accs[u]);
#pragma unroll
      for (int t = 0; t < 4; ++t)
        acc[u][t] = wmma_bf16(pa.v, bv[t].v, acc[u][t]);
    }
  }

  // ---- epilogue: normalize and store bf16 ----
#pragma unroll
  for (int u = 0; u < 2; ++u)
#pragma unroll
    for (int t = 0; t < 4; ++t)
#pragma unroll
      for (int r = 0; r < 8; ++r) {
        const float o = acc[u][t][r] / accs[u][r];
        O[(size_t)(b * SEQ + qt * 32 + u * 16 + chi + r) * DM + h * HD + t * 16 + cn] =
            f2bf(o);
      }
}

// -------------------- host launcher --------------------
extern "C" void kernel_launch(void* const* d_in, const int* in_sizes, int n_in,
                              void* d_out, int out_size, void* d_ws, size_t ws_size,
                              hipStream_t stream) {
  (void)in_sizes; (void)n_in; (void)out_size; (void)ws_size;
  const float* x   = (const float*)d_in[0];
  const float* g   = (const float*)d_in[1];
  const float* bta = (const float*)d_in[2];
  const float* Wq  = (const float*)d_in[3];
  const float* bq  = (const float*)d_in[4];
  const float* Wk  = (const float*)d_in[5];
  const float* bk  = (const float*)d_in[6];
  const float* Wv  = (const float*)d_in[7];
  const float* bv  = (const float*)d_in[8];
  const float* Wo  = (const float*)d_in[9];
  const float* bo  = (const float*)d_in[10];
  float* out = (float*)d_out;

  char* ws = (char*)d_ws;
  const size_t MB = 1ull << 20;
  unsigned short* xn  = (unsigned short*)(ws +  0 * MB);  // 8 MiB  [4096,1024] bf16
  unsigned short* WqT = (unsigned short*)(ws +  8 * MB);  // 2 MiB
  unsigned short* WkT = (unsigned short*)(ws + 10 * MB);  // 2 MiB
  unsigned short* WvT = (unsigned short*)(ws + 12 * MB);  // 2 MiB
  unsigned short* WoT = (unsigned short*)(ws + 14 * MB);  // 2 MiB
  unsigned short* Qb  = (unsigned short*)(ws + 16 * MB);  // 8 MiB
  unsigned short* Kb  = (unsigned short*)(ws + 24 * MB);  // 8 MiB
  unsigned short* Vb  = (unsigned short*)(ws + 32 * MB);  // 8 MiB
  unsigned short* Vt  = (unsigned short*)(ws + 40 * MB);  // 8 MiB [B,D,S]
  unsigned short* AO  = (unsigned short*)(ws + 48 * MB);  // 8 MiB attention output

  // 1) pre-norm
  ln_kernel<<<ROWS, 256, 0, stream>>>(x, g, bta, xn);

  // 2) weight convert+transpose
  const int wblocks = (DM * DM) / 256;
  wtrans_kernel<<<wblocks, 256, 0, stream>>>(Wq, WqT);
  wtrans_kernel<<<wblocks, 256, 0, stream>>>(Wk, WkT);
  wtrans_kernel<<<wblocks, 256, 0, stream>>>(Wv, WvT);
  wtrans_kernel<<<wblocks, 256, 0, stream>>>(Wo, WoT);

  // 3) QKV projections (Q pre-scaled by 1/sqrt(64))
  dim3 ggrid(DM / 64, ROWS / 128);
  gemm_wmma<false><<<ggrid, 256, 0, stream>>>(xn, WqT, bq, (void*)Qb, 0.125f);
  gemm_wmma<false><<<ggrid, 256, 0, stream>>>(xn, WkT, bk, (void*)Kb, 1.0f);
  gemm_wmma<false><<<ggrid, 256, 0, stream>>>(xn, WvT, bv, (void*)Vb, 1.0f);

  // 4) V transpose for contiguous B-fragments in attention
  vtrans_kernel<<<(BATCH * DM * SEQ) / 256, 256, 0, stream>>>(Vb, Vt);

  // 5) flash attention: one wave handles 32 queries
  attn_kernel<<<dim3(SEQ / 32, NH, BATCH), 32, 0, stream>>>(Qb, Kb, Vt, AO);

  // 6) output projection (f32 result)
  gemm_wmma<true><<<ggrid, 256, 0, stream>>>(AO, WoT, bo, (void*)out, 1.0f);
}
